// MaskedMultiHeadAttention_13649406066957
// MI455X (gfx1250) — matrix operand order repaired and checked by reading, not compiled, not run on hardware
//
#include <hip/hip_runtime.h>
#include <hip/hip_bf16.h>

typedef __bf16 bf16;
typedef __attribute__((ext_vector_type(16))) __bf16 v16bf;
typedef __attribute__((ext_vector_type(8)))  float  v8f;
typedef __attribute__((ext_vector_type(4)))  unsigned int u32x4;
typedef __attribute__((ext_vector_type(4)))  float  f32x4;

#define DMODEL 1024
#define NHEAD  16
#define HD     64
#define SEQ    2048
#define NB     4
#define NTOK   (NB * SEQ)   // 8192

union Frag { v16bf v; u32x4 q[2]; };

// ---------------------------------------------------------------------------
// Prep: fp32 -> bf16 conversions / weight transposes
// ---------------------------------------------------------------------------
__global__ void cvt_x_kernel(const float* __restrict__ x, bf16* __restrict__ xb) {
    int i = blockIdx.x * blockDim.x + threadIdx.x;          // 8192*1024 threads
    xb[i] = (bf16)x[i];
}

__global__ void tr_wqkv_kernel(const float* __restrict__ wq, const float* __restrict__ wk,
                               const float* __restrict__ wv, bf16* __restrict__ wt) {
    int i = blockIdx.x * blockDim.x + threadIdx.x;          // 3072*1024 threads
    int n = i >> 10, k = i & 1023;
    int proj = n >> 10, col = n & 1023;
    const float* w = (proj == 0) ? wq : (proj == 1) ? wk : wv;
    wt[i] = (bf16)w[k * 1024 + col];                        // Wt[n][k] = W[k][n]
}

__global__ void tr_wo_kernel(const float* __restrict__ wo, bf16* __restrict__ wt) {
    int i = blockIdx.x * blockDim.x + threadIdx.x;          // 1024*1024 threads
    int n = i >> 10, k = i & 1023;
    wt[i] = (bf16)wo[k * 1024 + n];
}

// ---------------------------------------------------------------------------
// QKV GEMM: [8192,1024] x Wt[3072,1024]^T -> q/k row-major, v transposed, bf16
// wave computes 32x32 output; block = 4 waves = 64x64
// ---------------------------------------------------------------------------
__global__ __launch_bounds__(128)
void qkv_gemm_kernel(const bf16* __restrict__ xb, const bf16* __restrict__ wt,
                     const float* __restrict__ bq, const float* __restrict__ bk,
                     const float* __restrict__ bv,
                     bf16* __restrict__ qb, bf16* __restrict__ kb, bf16* __restrict__ vt)
{
    const int lane = threadIdx.x & 31;
    const int w    = threadIdx.x >> 5;
    const int m0   = blockIdx.x * 64 + (w & 1) * 32;
    const int n0   = blockIdx.y * 64 + (w >> 1) * 32;
    const int lr   = lane & 15;
    const int kg   = lane >> 4;

    v8f acc[2][2] = {};
    for (int kk = 0; kk < DMODEL; kk += 32) {
        Frag a[2], bb[2];
#pragma unroll
        for (int t = 0; t < 2; ++t) {
            const bf16* pa = xb + (size_t)(m0 + t * 16 + lr) * DMODEL + kk + kg * 8;
            a[t].q[0] = *(const u32x4*)pa;
            a[t].q[1] = *(const u32x4*)(pa + 16);
            const bf16* pb = wt + (size_t)(n0 + t * 16 + lr) * DMODEL + kk + kg * 8;
            bb[t].q[0] = *(const u32x4*)pb;
            bb[t].q[1] = *(const u32x4*)(pb + 16);
        }
#pragma unroll
        for (int mt = 0; mt < 2; ++mt)
#pragma unroll
            for (int nt = 0; nt < 2; ++nt)
                acc[mt][nt] = __builtin_amdgcn_wmma_f32_16x16x32_bf16(
                    false, a[mt].v, false, bb[nt].v, (short)0, acc[mt][nt], false, false);
    }

#pragma unroll
    for (int nt = 0; nt < 2; ++nt) {
        int n = n0 + nt * 16 + lr;
        int proj = n >> 10;
        int col  = n & 1023;
        int h = col >> 6, d = col & 63;
        const float* bias = (proj == 0) ? bq : (proj == 1) ? bk : bv;
        float bias_v = bias[col];
#pragma unroll
        for (int mt = 0; mt < 2; ++mt) {
#pragma unroll
            for (int r = 0; r < 8; ++r) {
                int tok  = m0 + mt * 16 + r + kg * 8;
                int bidx = tok >> 11, s = tok & 2047;
                bf16 val = (bf16)(acc[mt][nt][r] + bias_v);
                if (proj == 0)
                    qb[(((size_t)(bidx * NHEAD + h)) * SEQ + s) * HD + d] = val;
                else if (proj == 1)
                    kb[(((size_t)(bidx * NHEAD + h)) * SEQ + s) * HD + d] = val;
                else
                    vt[(((size_t)(bidx * NHEAD + h)) * HD + d) * SEQ + s] = val;
            }
        }
    }
}

// ---------------------------------------------------------------------------
// Causal flash attention: wave = one 16-query tile; 32 keys / step
// ---------------------------------------------------------------------------
__global__ __launch_bounds__(128)
void attn_kernel(const bf16* __restrict__ qb, const bf16* __restrict__ kb,
                 const bf16* __restrict__ vt, bf16* __restrict__ ob)
{
    __shared__ bf16 lds_p[4][16 * 32];
    const int lane = threadIdx.x & 31;
    const int w    = threadIdx.x >> 5;
    const int lr   = lane & 15;
    const int kg   = lane >> 4;
    const int bh   = blockIdx.x;
    const int bI   = bh >> 4, h = bh & 15;
    const int q0   = (blockIdx.y * 4 + w) * 16;

    const bf16* Q = qb + (size_t)bh * SEQ * HD;
    const bf16* K = kb + (size_t)bh * SEQ * HD;
    const bf16* V = vt + (size_t)bh * HD * SEQ;

    Frag qa[2];
#pragma unroll
    for (int c = 0; c < 2; ++c) {
        const bf16* p = Q + (size_t)(q0 + lr) * HD + c * 32 + kg * 8;
        qa[c].q[0] = *(const u32x4*)p;
        qa[c].q[1] = *(const u32x4*)(p + 16);
    }

    v8f o[4] = {};
    float mrow[8], lsum[8];
#pragma unroll
    for (int r = 0; r < 8; ++r) { mrow[r] = -1e30f; lsum[r] = 0.f; }

    const int nk = (q0 + 47) >> 5;
    bf16* myP = lds_p[w];

    for (int kt = 0; kt < nk; ++kt) {
        const int k0 = kt * 32;
        v8f sc[2];
#pragma unroll
        for (int j = 0; j < 2; ++j) {
            v8f s = {};
#pragma unroll
            for (int c = 0; c < 2; ++c) {
                Frag kf;
                const bf16* p = K + (size_t)(k0 + j * 16 + lr) * HD + c * 32 + kg * 8;
                kf.q[0] = *(const u32x4*)p;
                kf.q[1] = *(const u32x4*)(p + 16);
                s = __builtin_amdgcn_wmma_f32_16x16x32_bf16(
                        false, qa[c].v, false, kf.v, (short)0, s, false, false);
            }
            sc[j] = s;
        }
        // scale + causal mask
#pragma unroll
        for (int j = 0; j < 2; ++j) {
            int key = k0 + j * 16 + lr;
#pragma unroll
            for (int r = 0; r < 8; ++r) {
                int row = q0 + r + kg * 8;
                float vv = sc[j][r] * 0.125f;            // 1/sqrt(64)
                sc[j][r] = (key > row) ? -1e30f : vv;
            }
        }
        // online softmax (row stats live per-lane-half, matching C layout)
#pragma unroll
        for (int r = 0; r < 8; ++r) {
            float tmax = fmaxf(sc[0][r], sc[1][r]);
            tmax = fmaxf(tmax, __shfl_xor(tmax, 1));
            tmax = fmaxf(tmax, __shfl_xor(tmax, 2));
            tmax = fmaxf(tmax, __shfl_xor(tmax, 4));
            tmax = fmaxf(tmax, __shfl_xor(tmax, 8));
            float Mn    = fmaxf(mrow[r], tmax);
            float alpha = __expf(mrow[r] - Mn);
            float p0 = __expf(sc[0][r] - Mn);
            float p1 = __expf(sc[1][r] - Mn);
            sc[0][r] = p0; sc[1][r] = p1;
            float ps = p0 + p1;
            ps += __shfl_xor(ps, 1);
            ps += __shfl_xor(ps, 2);
            ps += __shfl_xor(ps, 4);
            ps += __shfl_xor(ps, 8);
            lsum[r] = lsum[r] * alpha + ps;
            mrow[r] = Mn;
#pragma unroll
            for (int t = 0; t < 4; ++t) o[t][r] *= alpha;
        }
        // stage P in LDS (row-major [16 q][32 k]) to re-layout as A fragment
#pragma unroll
        for (int j = 0; j < 2; ++j)
#pragma unroll
            for (int r = 0; r < 8; ++r)
                myP[(r + kg * 8) * 32 + j * 16 + lr] = (bf16)sc[j][r];

        Frag pa;
        {
            const bf16* p = myP + lr * 32 + kg * 8;     // per-wave LDS ops are in-order
            pa.q[0] = *(const u32x4*)p;
            pa.q[1] = *(const u32x4*)(p + 16);
        }
        // O += P @ V   (Vt[n=d][k=key] is a direct B-fragment)
#pragma unroll
        for (int t = 0; t < 4; ++t) {
            Frag vf;
            const bf16* p = V + (size_t)(t * 16 + lr) * SEQ + k0 + kg * 8;
            vf.q[0] = *(const u32x4*)p;
            vf.q[1] = *(const u32x4*)(p + 16);
            o[t] = __builtin_amdgcn_wmma_f32_16x16x32_bf16(
                       false, pa.v, false, vf.v, (short)0, o[t], false, false);
        }
    }

#pragma unroll
    for (int r = 0; r < 8; ++r) {
        int s = q0 + r + kg * 8;
        float inv = 1.f / lsum[r];
#pragma unroll
        for (int t = 0; t < 4; ++t)
            ob[((size_t)(bI * SEQ + s)) * DMODEL + h * HD + t * 16 + lr] =
                (bf16)(o[t][r] * inv);
    }
}

// ---------------------------------------------------------------------------
// Output projection + bias + residual -> fp32 y
// ---------------------------------------------------------------------------
__global__ __launch_bounds__(128)
void oproj_kernel(const bf16* __restrict__ ob, const bf16* __restrict__ wot,
                  const float* __restrict__ bo, const float* __restrict__ x,
                  float* __restrict__ y)
{
    const int lane = threadIdx.x & 31;
    const int w    = threadIdx.x >> 5;
    const int m0   = blockIdx.x * 64 + (w & 1) * 32;
    const int n0   = blockIdx.y * 64 + (w >> 1) * 32;
    const int lr   = lane & 15;
    const int kg   = lane >> 4;

    v8f acc[2][2] = {};
    for (int kk = 0; kk < DMODEL; kk += 32) {
        Frag a[2], bb[2];
#pragma unroll
        for (int t = 0; t < 2; ++t) {
            const bf16* pa = ob + (size_t)(m0 + t * 16 + lr) * DMODEL + kk + kg * 8;
            a[t].q[0] = *(const u32x4*)pa;
            a[t].q[1] = *(const u32x4*)(pa + 16);
            const bf16* pb = wot + (size_t)(n0 + t * 16 + lr) * DMODEL + kk + kg * 8;
            bb[t].q[0] = *(const u32x4*)pb;
            bb[t].q[1] = *(const u32x4*)(pb + 16);
        }
#pragma unroll
        for (int mt = 0; mt < 2; ++mt)
#pragma unroll
            for (int nt = 0; nt < 2; ++nt)
                acc[mt][nt] = __builtin_amdgcn_wmma_f32_16x16x32_bf16(
                    false, a[mt].v, false, bb[nt].v, (short)0, acc[mt][nt], false, false);
    }

#pragma unroll
    for (int nt = 0; nt < 2; ++nt) {
        int n = n0 + nt * 16 + lr;
        float bias_v = bo[n];
#pragma unroll
        for (int mt = 0; mt < 2; ++mt)
#pragma unroll
            for (int r = 0; r < 8; ++r) {
                size_t tok = (size_t)(m0 + mt * 16 + r + kg * 8);
                y[tok * DMODEL + n] = acc[mt][nt][r] + bias_v + x[tok * DMODEL + n];
            }
    }
}

// ---------------------------------------------------------------------------
// LayerNorm: one wave per 1024-element row
// ---------------------------------------------------------------------------
__global__ __launch_bounds__(128)
void ln_kernel(const float* __restrict__ y, const float* __restrict__ g,
               const float* __restrict__ bta, float* __restrict__ out)
{
    const int lane = threadIdx.x & 31;
    const int w    = threadIdx.x >> 5;
    const int row  = blockIdx.x * 4 + w;

    const f32x4* p = (const f32x4*)(y + (size_t)row * DMODEL);
    f32x4 vals[8];
    float sum = 0.f, sq = 0.f;
#pragma unroll
    for (int i = 0; i < 8; ++i) {
        f32x4 v = p[lane + i * 32];
        vals[i] = v;
        sum += v[0] + v[1] + v[2] + v[3];
        sq  += v[0]*v[0] + v[1]*v[1] + v[2]*v[2] + v[3]*v[3];
    }
#pragma unroll
    for (int m = 16; m >= 1; m >>= 1) {
        sum += __shfl_xor(sum, m);
        sq  += __shfl_xor(sq,  m);
    }
    float mu   = sum * (1.f / 1024.f);
    float var  = sq * (1.f / 1024.f) - mu * mu;
    float rstd = rsqrtf(var + 1e-5f);

    const f32x4* pg = (const f32x4*)g;
    const f32x4* pb = (const f32x4*)bta;
    f32x4* po = (f32x4*)(out + (size_t)row * DMODEL);
#pragma unroll
    for (int i = 0; i < 8; ++i) {
        f32x4 gv = pg[lane + i * 32], bv = pb[lane + i * 32], v = vals[i], r;
#pragma unroll
        for (int e = 0; e < 4; ++e)
            r[e] = (v[e] - mu) * rstd * gv[e] + bv[e];
        po[lane + i * 32] = r;
    }
}

// ---------------------------------------------------------------------------
extern "C" void kernel_launch(void* const* d_in, const int* in_sizes, int n_in,
                              void* d_out, int out_size, void* d_ws, size_t ws_size,
                              hipStream_t stream) {
    const float* x  = (const float*)d_in[0];
    const float* wq = (const float*)d_in[1];
    const float* bq = (const float*)d_in[2];
    const float* wk = (const float*)d_in[3];
    const float* bk = (const float*)d_in[4];
    const float* wv = (const float*)d_in[5];
    const float* bv = (const float*)d_in[6];
    const float* wo = (const float*)d_in[7];
    const float* bo = (const float*)d_in[8];
    const float* lg = (const float*)d_in[9];
    const float* lb = (const float*)d_in[10];
    float* out = (float*)d_out;

    char* ws = (char*)d_ws;
    size_t off = 0;
    auto take = [&](size_t bytes) -> char* {
        char* p = ws + off;
        off = (off + bytes + 255) & ~(size_t)255;
        return p;
    };
    bf16* xb    = (bf16*)take((size_t)NTOK * DMODEL * 2);          // 16 MB
    bf16* wqkvt = (bf16*)take((size_t)3 * DMODEL * DMODEL * 2);    //  6 MB
    bf16* wot   = (bf16*)take((size_t)DMODEL * DMODEL * 2);        //  2 MB
    bf16* qb    = (bf16*)take((size_t)NTOK * DMODEL * 2);          // 16 MB
    bf16* kb    = (bf16*)take((size_t)NTOK * DMODEL * 2);          // 16 MB
    bf16* vt    = (bf16*)take((size_t)NTOK * DMODEL * 2);          // 16 MB
    bf16* ob    = (bf16*)take((size_t)NTOK * DMODEL * 2);          // 16 MB
    float* yy   = (float*)take((size_t)NTOK * DMODEL * 4);         // 32 MB

    cvt_x_kernel  <<<NTOK * DMODEL / 256, 256, 0, stream>>>(x, xb);
    tr_wqkv_kernel<<<3 * DMODEL * DMODEL / 256, 256, 0, stream>>>(wq, wk, wv, wqkvt);
    tr_wo_kernel  <<<DMODEL * DMODEL / 256, 256, 0, stream>>>(wo, wot);

    qkv_gemm_kernel<<<dim3(NTOK / 64, 3 * DMODEL / 64), 128, 0, stream>>>(
        xb, wqkvt, bq, bk, bv, qb, kb, vt);

    attn_kernel<<<dim3(NB * NHEAD, SEQ / 16 / 4), 128, 0, stream>>>(qb, kb, vt, ob);

    oproj_kernel<<<dim3(NTOK / 64, DMODEL / 64), 128, 0, stream>>>(ob, wot, bo, x, yy);

    ln_kernel<<<NTOK / 4, 128, 0, stream>>>(yy, lg, lb, out);
}
